// QANet_67104569032834
// MI455X (gfx1250) — compile-verified, matrix operand order checked by reading, not run on hardware
//
#include <hip/hip_runtime.h>
#include <hip/hip_bf16.h>
#include <math.h>

// ---------------- model constants ----------------
#define B_N   32
#define PARA  400
#define QUES  50
#define DM    128
#define NH    8
#define DEMB  500          // 300 word + 200 char
#define DWORD_E 300
#define DCHAR_E 200
#define CHLEN 16
#define NEGV  (-1e30f)

typedef __attribute__((ext_vector_type(16))) _Float16 v16h;
typedef __attribute__((ext_vector_type(8)))  float    v8f;

static inline unsigned nblk(long n) { return (unsigned)((n + 255) / 256); }

// =================================================================
// Generic strided-batched GEMM via WMMA f32<=f16 fragments.
// C[z] = act(alpha * A[z]*B[z] + bias)   with per-(batch,head) strides.
// Block = 128 threads = 4 waves; each wave computes a 16x32 C tile
// (two 16x16 WMMA accumulators sharing one A fragment).
// A: [M,K] ld lda (row major). B: row major [K,N] (transB=0) or [N,K] (transB=1).
// act: 0=none 1=relu 2=sigmoid
// All fragment loads are unconditional (clamped addresses + value select),
// so the K-loop is branch-free and EXEC stays all-ones around the WMMAs.
// =================================================================
__global__ void k_gemm_wmma(const float* __restrict__ A, const float* __restrict__ Bm,
                            float* __restrict__ C, const float* __restrict__ bias,
                            int M, int N, int K, int lda, int ldb, int ldc,
                            long sAb, long sAh, long sBb, long sBh, long sCb, long sCh,
                            int H, int transB, int act, float alpha)
{
    int z = blockIdx.z;
    int b = z / H, h = z % H;
    const float* Ab = A  + (long)b * sAb + (long)h * sAh;
    const float* Bb = Bm + (long)b * sBb + (long)h * sBh;
    float*       Cb = C  + (long)b * sCb + (long)h * sCh;

    int wid  = threadIdx.x >> 5;          // wave 0..3 -> M sub-tile
    int lane = threadIdx.x & 31;
    int grp  = lane >> 4;                 // 0/1
    int lmod = lane & 15;

    int tm = blockIdx.x * 64 + wid * 16;
    int tn = blockIdx.y * 32;

    // hoisted per-lane row/col bases (k-invariant)
    int m   = tm + lmod;
    int mc  = (m < M) ? m : (M - 1);
    bool mok = (m < M);
    const float* Arow = Ab + (long)mc * lda;

    int n0   = tn + lmod;
    int n0c  = (n0 < N) ? n0 : (N - 1);
    bool n0ok = (n0 < N);
    int n1   = tn + 16 + lmod;
    int n1c  = (n1 < N) ? n1 : (N - 1);
    bool n1ok = (n1 < N);

    long bstride = transB ? 1 : (long)ldb;   // element stride along K
    const float* Brow0 = transB ? (Bb + (long)n0c * ldb) : (Bb + n0c);
    const float* Brow1 = transB ? (Bb + (long)n1c * ldb) : (Bb + n1c);

    v8f acc0 = {}, acc1 = {};
    for (int k0 = 0; k0 < K; k0 += 32) {
        v16h af = {}, bf0 = {}, bf1 = {};
#pragma unroll
        for (int e = 0; e < 16; ++e) {
            int k = k0 + ((e >= 8) ? 16 : 0) + 8 * grp + (e & 7);
            int kc = (k < K) ? k : (K - 1);
            bool kok = (k < K);
            float av = Arow[kc];
            av = (mok && kok) ? av : 0.0f;
            af[e] = (_Float16)av;
            float bv0 = Brow0[(long)kc * bstride];
            bv0 = (n0ok && kok) ? bv0 : 0.0f;
            bf0[e] = (_Float16)bv0;
            float bv1 = Brow1[(long)kc * bstride];
            bv1 = (n1ok && kok) ? bv1 : 0.0f;
            bf1[e] = (_Float16)bv1;
        }
        acc0 = __builtin_amdgcn_wmma_f32_16x16x32_f16(false, af, false, bf0,
                                                      (short)0, acc0, false, false);
        acc1 = __builtin_amdgcn_wmma_f32_16x16x32_f16(false, af, false, bf1,
                                                      (short)0, acc1, false, false);
    }
#pragma unroll
    for (int r = 0; r < 8; ++r) {
        int mm = tm + r + 8 * grp;
        if (mm < M) {
            if (n0 < N) {
                float v = acc0[r] * alpha;
                if (bias) v += bias[n0];
                if (act == 1)      v = fmaxf(v, 0.0f);
                else if (act == 2) v = 1.0f / (1.0f + expf(-v));
                Cb[(long)mm * ldc + n0] = v;
            }
            if (n1 < N) {
                float v = acc1[r] * alpha;
                if (bias) v += bias[n1];
                if (act == 1)      v = fmaxf(v, 0.0f);
                else if (act == 2) v = 1.0f / (1.0f + expf(-v));
                Cb[(long)mm * ldc + n1] = v;
            }
        }
    }
}

// ---------------- elementwise / small kernels ----------------
__global__ void k_mask(const int* __restrict__ w, float* __restrict__ m, long n) {
    long i = (long)blockIdx.x * 256 + threadIdx.x;
    if (i < n) m[i] = (w[i] != 0) ? 1.0f : 0.0f;
}

__global__ void k_embed(const int* __restrict__ wids, const int* __restrict__ cids,
                        const float* __restrict__ wemb, const float* __restrict__ cemb,
                        float* __restrict__ out, long ntok)
{
    long total = ntok * DEMB;
    long i = (long)blockIdx.x * 256 + threadIdx.x;
    if (i >= total) return;
    long tok = i / DEMB; int d = (int)(i % DEMB);
    if (d < DWORD_E) {
        out[i] = wemb[(long)wids[tok] * DWORD_E + d];
    } else {
        int dd = d - DWORD_E;
        const int* cp = cids + tok * CHLEN;
        float m = cemb[(long)cp[0] * DCHAR_E + dd];
        for (int t = 1; t < CHLEN; ++t)
            m = fmaxf(m, cemb[(long)cp[t] * DCHAR_E + dd]);
        out[i] = m;
    }
}

__global__ void k_hw(float* __restrict__ x, const float* __restrict__ h,
                     const float* __restrict__ t, long n)
{
    long i = (long)blockIdx.x * 256 + threadIdx.x;
    if (i < n) { float tv = t[i]; x[i] = h[i] * tv + x[i] * (1.0f - tv); }
}

__global__ void k_pos_add(float* __restrict__ x, int L, int D, long total) {
    long i = (long)blockIdx.x * 256 + threadIdx.x;
    if (i >= total) return;
    int d = (int)(i % D);
    long bl = i / D;
    int l = (int)(bl % L);
    float fD = (float)D;
    float freq, phase;
    if ((d & 1) == 0) { freq = powf(10000.0f, -(float)d / fD); phase = 0.0f; }
    else              { freq = -powf(10000.0f, (1.0f - (float)d) / fD); phase = 1.57079632679f; }
    x[i] += sinf((float)l * freq + phase);
}

__global__ void k_dwconv(const float* __restrict__ x, const float* __restrict__ dw,
                         const float* __restrict__ db, float* __restrict__ y,
                         int L, int Cch, int ksz, int relu, long total)
{
    long i = (long)blockIdx.x * 256 + threadIdx.x;
    if (i >= total) return;
    int c = (int)(i % Cch);
    long bl = i / Cch;
    int l = (int)(bl % L);
    long b = bl / L;
    int pad = (ksz - 1) / 2;
    float acc = db[c];
    for (int j = 0; j < ksz; ++j) {
        int ll = l + j - pad;
        if (ll >= 0 && ll < L)
            acc += x[((long)b * L + ll) * Cch + c] * dw[c * ksz + j];
    }
    if (relu) acc = fmaxf(acc, 0.0f);
    y[i] = acc;
}

// out[b] = base[b] + LayerNorm2D(res[b]) * w + bb   (norm over L*C elems)
__global__ void k_ln2_add(float* __restrict__ out, const float* __restrict__ base,
                          const float* __restrict__ res, const float* __restrict__ w,
                          const float* __restrict__ bb, int n)
{
    __shared__ float r1[256], r2[256];
    long off = (long)blockIdx.x * n;
    int t = threadIdx.x;
    float s = 0.0f, s2 = 0.0f;
    for (int i = t; i < n; i += 256) { float v = res[off + i]; s += v; s2 += v * v; }
    r1[t] = s; r2[t] = s2; __syncthreads();
    for (int o = 128; o > 0; o >>= 1) {
        if (t < o) { r1[t] += r1[t + o]; r2[t] += r2[t + o]; }
        __syncthreads();
    }
    float mu = r1[0] / (float)n;
    float var = r2[0] / (float)n - mu * mu;
    float rstd = rsqrtf(var + 1e-5f);
    for (int i = t; i < n; i += 256)
        out[off + i] = base[off + i] + (res[off + i] - mu) * rstd * w[i] + bb[i];
}

// pack wq[H,128,16] -> W[128(K),128(N=h*16+e)]
__global__ void k_pack_w(const float* __restrict__ w, float* __restrict__ o) {
    int i = blockIdx.x * 256 + threadIdx.x;
    if (i >= DM * DM) return;
    int k = i >> 7, n = i & 127;
    o[i] = w[(n >> 4) * (DM * 16) + k * 16 + (n & 15)];
}

// attention masked softmax, in-place on S [B,H,L,L]; one block per row
__global__ void k_attn_softmax(float* __restrict__ S, const float* __restrict__ mask,
                               int H, int L)
{
    __shared__ float red[256];
    long line = blockIdx.x;
    int b = (int)(line / ((long)H * L));
    int l = (int)(line % L);
    float* row = S + line * L;
    float rm = mask[(long)b * L + l];
    const float* cm = mask + (long)b * L;
    int t = threadIdx.x;
    float mx = -3.4e38f;
    for (int j = t; j < L; j += 256) {
        float m = rm * cm[j];
        float v = row[j] * m + NEGV * (1.0f - m);
        row[j] = v; mx = fmaxf(mx, v);
    }
    red[t] = mx; __syncthreads();
    for (int o = 128; o > 0; o >>= 1) { if (t < o) red[t] = fmaxf(red[t], red[t + o]); __syncthreads(); }
    mx = red[0]; __syncthreads();
    float sm = 0.0f;
    for (int j = t; j < L; j += 256) { float e = expf(row[j] - mx); row[j] = e; sm += e; }
    red[t] = sm; __syncthreads();
    for (int o = 128; o > 0; o >>= 1) { if (t < o) red[t] += red[t + o]; __syncthreads(); }
    float inv = 1.0f / red[0];
    for (int j = t; j < L; j += 256) row[j] *= inv;
}

// softmax over last dim with column mask (CQ S1): in [B,R,Cn] -> out
__global__ void k_softmax_rows(const float* __restrict__ in, float* __restrict__ out,
                               const float* __restrict__ cmask, int R, int Cn)
{
    __shared__ float red[256];
    long line = blockIdx.x;           // b*R + r
    int b = (int)(line / R);
    const float* src = in + line * Cn;
    float* dst = out + line * Cn;
    const float* cm = cmask + (long)b * Cn;
    int t = threadIdx.x;
    float mx = -3.4e38f;
    for (int j = t; j < Cn; j += 256) {
        float m = cm[j];
        float v = src[j] * m + NEGV * (1.0f - m);
        dst[j] = v; mx = fmaxf(mx, v);
    }
    red[t] = mx; __syncthreads();
    for (int o = 128; o > 0; o >>= 1) { if (t < o) red[t] = fmaxf(red[t], red[t + o]); __syncthreads(); }
    mx = red[0]; __syncthreads();
    float sm = 0.0f;
    for (int j = t; j < Cn; j += 256) { float e = expf(dst[j] - mx); dst[j] = e; sm += e; }
    red[t] = sm; __syncthreads();
    for (int o = 128; o > 0; o >>= 1) { if (t < o) red[t] += red[t + o]; __syncthreads(); }
    float inv = 1.0f / red[0];
    for (int j = t; j < Cn; j += 256) dst[j] *= inv;
}

// softmax over axis1 (rows) with row mask (CQ S2): in [B,R,Cn], line = b*Cn + m
__global__ void k_softmax_axis1(const float* __restrict__ in, float* __restrict__ out,
                                const float* __restrict__ rmask, int R, int Cn)
{
    __shared__ float red[256];
    long line = blockIdx.x;
    int b = (int)(line / Cn);
    int m = (int)(line % Cn);
    const float* src = in + (long)b * R * Cn + m;
    float* dst = out + (long)b * R * Cn + m;
    const float* rm = rmask + (long)b * R;
    int t = threadIdx.x;
    float mx = -3.4e38f;
    for (int l = t; l < R; l += 256) {
        float mm = rm[l];
        float v = src[(long)l * Cn] * mm + NEGV * (1.0f - mm);
        dst[(long)l * Cn] = v; mx = fmaxf(mx, v);
    }
    red[t] = mx; __syncthreads();
    for (int o = 128; o > 0; o >>= 1) { if (t < o) red[t] = fmaxf(red[t], red[t + o]); __syncthreads(); }
    mx = red[0]; __syncthreads();
    float sm = 0.0f;
    for (int l = t; l < R; l += 256) { float e = expf(dst[(long)l * Cn] - mx); dst[(long)l * Cn] = e; sm += e; }
    red[t] = sm; __syncthreads();
    for (int o = 128; o > 0; o >>= 1) { if (t < o) red[t] += red[t + o]; __syncthreads(); }
    float inv = 1.0f / red[0];
    for (int l = t; l < R; l += 256) dst[(long)l * Cn] *= inv;
}

__global__ void k_scale_cols(const float* __restrict__ x, const float* __restrict__ w,
                             float* __restrict__ y, long n)
{
    long i = (long)blockIdx.x * 256 + threadIdx.x;
    if (i < n) y[i] = x[i] * w[i % DM];
}

__global__ void k_dotw(const float* __restrict__ x, const float* __restrict__ w,
                       float* __restrict__ o, long rows, int d)
{
    long r = (long)blockIdx.x * 256 + threadIdx.x;
    if (r >= rows) return;
    const float* xp = x + r * d;
    float acc = 0.0f;
    for (int i = 0; i < d; ++i) acc += xp[i] * w[i];
    o[r] = acc;
}

__global__ void k_addS(float* __restrict__ S, const float* __restrict__ s0,
                       const float* __restrict__ s1, const float* __restrict__ bias,
                       int R, int Cn, long total)
{
    long i = (long)blockIdx.x * 256 + threadIdx.x;
    if (i >= total) return;
    int m = (int)(i % Cn);
    long bl = i / Cn;
    int l = (int)(bl % R);
    int b = (int)(bl / R);
    S[i] += s0[(long)b * R + l] + s1[(long)b * Cn + m] + bias[0];
}

__global__ void k_concat_cq(const float* __restrict__ C, const float* __restrict__ A,
                            const float* __restrict__ Bv, float* __restrict__ out, long total)
{
    long i = (long)blockIdx.x * 256 + threadIdx.x;
    if (i >= total) return;
    int d = (int)(i % (4 * DM));
    long bl = i / (4 * DM);
    long base = bl * DM;
    float v;
    if (d < DM)            v = C[base + d];
    else if (d < 2 * DM)   v = A[base + d - DM];
    else if (d < 3 * DM)   v = C[base + d - 2 * DM] * A[base + d - 2 * DM];
    else                   v = C[base + d - 3 * DM] * Bv[base + d - 3 * DM];
    out[i] = v;
}

__global__ void k_ptr(const float* __restrict__ Ma, const float* __restrict__ Mb,
                      const float* __restrict__ w, const float* __restrict__ bsc,
                      float* __restrict__ p, long rows, int d)
{
    long r = (long)blockIdx.x * 256 + threadIdx.x;
    if (r >= rows) return;
    const float* a = Ma + r * d;
    const float* b = Mb + r * d;
    float acc = bsc[0];
    for (int i = 0; i < d; ++i) acc += a[i] * w[i] + b[i] * w[d + i];
    p[r] = acc;
}

__global__ void k_logsoftmax(const float* __restrict__ x, const float* __restrict__ mask,
                             float* __restrict__ out, int L)
{
    __shared__ float red[256];
    int b = blockIdx.x;
    const float* xp = x + (long)b * L;
    const float* mp = mask + (long)b * L;
    float* op = out + (long)b * L;
    int t = threadIdx.x;
    float mx = -3.4e38f;
    for (int j = t; j < L; j += 256) {
        float m = mp[j];
        float v = xp[j] * m + NEGV * (1.0f - m);
        op[j] = v; mx = fmaxf(mx, v);
    }
    red[t] = mx; __syncthreads();
    for (int o = 128; o > 0; o >>= 1) { if (t < o) red[t] = fmaxf(red[t], red[t + o]); __syncthreads(); }
    mx = red[0]; __syncthreads();
    float sm = 0.0f;
    for (int j = t; j < L; j += 256) sm += expf(op[j] - mx);
    red[t] = sm; __syncthreads();
    for (int o = 128; o > 0; o >>= 1) { if (t < o) red[t] += red[t + o]; __syncthreads(); }
    float lse = mx + logf(red[0]);
    for (int j = t; j < L; j += 256) op[j] = op[j] - lse;
}

// =================================================================
// host side
// =================================================================
struct Enc {
    const float *bk, *bq, *bv, *fc_b, *fc_w, *wk, *wq, *wv;
    const float *db[3], *dw[3], *pb[3], *pw[3];
    const float *nAw, *nAb, *nOw, *nOb;
    const float *nCw[2], *nCb[2];
    const float *out_b, *out_w;
    int nconv, ksz, in_ch, L;
};

static Enc make_enc(void* const* d_in, int base, int nconv, int ksz, int in_ch, int L) {
    Enc e{};
    auto f = [&](int o) { return (const float*)d_in[base + o]; };
    e.bk = f(0); e.bq = f(1); e.bv = f(2); e.fc_b = f(3); e.fc_w = f(4);
    e.wk = f(5); e.wq = f(6); e.wv = f(7);
    int idx = 8;
    for (int i = 0; i < nconv - 1; ++i) {
        e.db[i] = f(idx++); e.dw[i] = f(idx++); e.pb[i] = f(idx++); e.pw[i] = f(idx++);
    }
    e.nAw = f(idx++); e.nAb = f(idx++); e.nOw = f(idx++); e.nOb = f(idx++);
    for (int i = 0; i < nconv - 2; ++i) { e.nCw[i] = f(idx++); e.nCb[i] = f(idx++); }
    e.out_b = f(idx++); e.out_w = f(idx++);
    e.nconv = nconv; e.ksz = ksz; e.in_ch = in_ch; e.L = L;
    return e;
}

struct WS {
    float *uni;                       // 40,960,000 floats, phase-reused
    float *cmask, *qmask, *Qemb, *C128, *Q128;
    float *qb, *kb, *vb, *ob, *tmpB, *tmpc;
    float *wqp, *wkp, *wvp;
    float *M1, *M2, *M3, *cq512;
};

static void gemm_full(hipStream_t s, const float* A, const float* Bm, float* C,
                      const float* bias, int M, int N, int K,
                      int lda, int ldb, int ldc,
                      long sAb, long sAh, long sBb, long sBh, long sCb, long sCh,
                      int H, int nz, int transB, int act, float alpha)
{
    dim3 g((M + 63) / 64, (N + 31) / 32, nz);
    k_gemm_wmma<<<g, dim3(128, 1, 1), 0, s>>>(A, Bm, C, bias, M, N, K, lda, ldb, ldc,
                                              sAb, sAh, sBb, sBh, sCb, sCh,
                                              H, transB, act, alpha);
}

static void gemm2(hipStream_t s, const float* A, const float* Bm, float* C,
                  const float* bias, int M, int N, int K, int transB, int act)
{
    gemm_full(s, A, Bm, C, bias, M, N, K, K, transB ? K : N, N,
              0, 0, 0, 0, 0, 0, 1, 1, transB, act, 1.0f);
}

static void run_enc(hipStream_t s, const Enc& e, const float* mask,
                    float* xin, float* xout, WS& w)
{
    int L = e.L, IC = e.in_ch;
    long tot_in = (long)B_N * L * IC;
    long n128 = (long)B_N * L * DM;

    k_pos_add<<<nblk(tot_in), 256, 0, s>>>(xin, L, IC, tot_in);

    float* dwout = (IC > DM) ? (w.uni + 6400000) : w.tmpc;
    k_dwconv<<<nblk(tot_in), 256, 0, s>>>(xin, e.dw[0], e.db[0], dwout, L, IC, e.ksz, 1, tot_in);
    float* x = xout;
    gemm2(s, dwout, e.pw[0], x, e.pb[0], B_N * L, DM, IC, 1, 1);

    for (int i = 1; i < e.nconv - 1; ++i) {
        k_dwconv<<<nblk(n128), 256, 0, s>>>(x, e.dw[i], e.db[i], w.tmpc, L, DM, e.ksz, 1, n128);
        gemm2(s, w.tmpc, e.pw[i], w.tmpB, e.pb[i], B_N * L, DM, DM, 1, 1);
        k_ln2_add<<<B_N, 256, 0, s>>>(x, w.tmpB, x, e.nCw[i - 1], e.nCb[i - 1], L * DM);
    }

    // --- self attention ---
    k_pack_w<<<64, 256, 0, s>>>(e.wq, w.wqp);
    k_pack_w<<<64, 256, 0, s>>>(e.wk, w.wkp);
    k_pack_w<<<64, 256, 0, s>>>(e.wv, w.wvp);
    gemm2(s, x, w.wqp, w.qb, e.bq, B_N * L, DM, DM, 0, 0);
    gemm2(s, x, w.wkp, w.kb, e.bk, B_N * L, DM, DM, 0, 0);
    gemm2(s, x, w.wvp, w.vb, e.bv, B_N * L, DM, DM, 0, 0);

    float* S = w.uni;
    gemm_full(s, w.qb, w.kb, S, nullptr, L, L, 16,
              DM, DM, L,
              (long)L * DM, 16, (long)L * DM, 16, (long)NH * L * L, (long)L * L,
              NH, B_N * NH, 1, 0, 0.25f);
    k_attn_softmax<<<B_N * NH * L, 256, 0, s>>>(S, mask, NH, L);
    gemm_full(s, S, w.vb, w.ob, nullptr, L, 16, L,
              L, DM, DM,
              (long)NH * L * L, (long)L * L, (long)L * DM, 16, (long)L * DM, 16,
              NH, B_N * NH, 0, 0, 1.0f);
    gemm2(s, w.ob, e.fc_w, w.tmpB, e.fc_b, B_N * L, DM, DM, 1, 0);
    k_ln2_add<<<B_N, 256, 0, s>>>(x, w.tmpB, x, e.nAw, e.nAb, L * DM);

    // --- output projection ---
    gemm2(s, x, e.out_w, w.tmpB, e.out_b, B_N * L, DM, DM, 1, 0);
    k_ln2_add<<<B_N, 256, 0, s>>>(x, w.tmpB, x, e.nOw, e.nOb, L * DM);
}

extern "C" void kernel_launch(void* const* d_in, const int* in_sizes, int n_in,
                              void* d_out, int out_size, void* d_ws, size_t ws_size,
                              hipStream_t stream)
{
    (void)in_sizes; (void)n_in; (void)out_size; (void)ws_size;
    auto F = [&](int i) { return (const float*)d_in[i]; };
    auto Ii = [&](int i) { return (const int*)d_in[i]; };

    // pytree leaf order: sorted dict keys, ordered lists/tuples
    const float* char_emb = F(0);
    const float* cq_bias = F(1); const float* w4c = F(2);
    const float* w4mul = F(3);   const float* w4q = F(4);
    Enc ctx = make_enc(d_in, 5, 4, 7, DEMB, PARA);
    Enc dec[7];
    for (int i = 0; i < 7; ++i) dec[i] = make_enc(d_in, 35 + 18 * i, 2, 5, DM, PARA);
    const float* H_b = F(161); const float* H_w = F(162); const float* T_w = F(163);
    const float* b1 = F(164);  const float* b2 = F(165);
    const float* w1 = F(166);  const float* w2 = F(167);
    Enc qen = make_enc(d_in, 168, 4, 7, DEMB, QUES);
    const float* rdb = F(198); const float* rdw = F(199);
    const float* rpb = F(200); const float* rpw = F(201);
    const float* word_emb = F(202);
    const int* cw = Ii(203); const int* cc = Ii(204);
    const int* qw = Ii(205); const int* qc = Ii(206);

    // ------- workspace arena -------
    float* p = (float*)d_ws;
    auto alloc = [&](long n) { float* r = p; p += n; return r; };
    WS w{};
    w.uni   = alloc(40960000L);
    w.cmask = alloc((long)B_N * PARA);
    w.qmask = alloc((long)B_N * QUES);
    w.Qemb  = alloc((long)B_N * QUES * DEMB);
    w.C128  = alloc((long)B_N * PARA * DM);
    w.Q128  = alloc((long)B_N * QUES * DM);
    long nC128 = (long)B_N * PARA * DM;
    w.qb = alloc(nC128); w.kb = alloc(nC128); w.vb = alloc(nC128);
    w.ob = alloc(nC128); w.tmpB = alloc(nC128); w.tmpc = alloc(nC128);
    w.wqp = alloc(DM * DM); w.wkp = alloc(DM * DM); w.wvp = alloc(DM * DM);
    w.M1 = alloc(nC128); w.M2 = alloc(nC128); w.M3 = alloc(nC128);
    w.cq512 = alloc((long)B_N * PARA * 4 * DM);

    // ------- embeddings + masks -------
    long nCemb = (long)B_N * PARA * DEMB;
    long nQemb = (long)B_N * QUES * DEMB;
    float* Cemb = w.uni;                 // union phase 1
    float* hb = w.uni + 6400000;
    float* tb = w.uni + 12800000;
    k_mask<<<nblk(B_N * PARA), 256, 0, stream>>>(cw, w.cmask, (long)B_N * PARA);
    k_mask<<<nblk(B_N * QUES), 256, 0, stream>>>(qw, w.qmask, (long)B_N * QUES);
    k_embed<<<nblk(nCemb), 256, 0, stream>>>(cw, cc, word_emb, char_emb, Cemb, (long)B_N * PARA);
    k_embed<<<nblk(nQemb), 256, 0, stream>>>(qw, qc, word_emb, char_emb, w.Qemb, (long)B_N * QUES);

    // ------- highway (shared weights) -------
    gemm2(stream, Cemb, H_w, hb, H_b, B_N * PARA, DEMB, DEMB, 1, 1);
    gemm2(stream, Cemb, T_w, tb, nullptr, B_N * PARA, DEMB, DEMB, 1, 2);
    k_hw<<<nblk(nCemb), 256, 0, stream>>>(Cemb, hb, tb, nCemb);
    gemm2(stream, w.Qemb, H_w, hb, H_b, B_N * QUES, DEMB, DEMB, 1, 1);
    gemm2(stream, w.Qemb, T_w, tb, nullptr, B_N * QUES, DEMB, DEMB, 1, 2);
    k_hw<<<nblk(nQemb), 256, 0, stream>>>(w.Qemb, hb, tb, nQemb);

    // ------- encoder blocks -------
    run_enc(stream, ctx, w.cmask, Cemb, w.C128, w);
    run_enc(stream, qen, w.qmask, w.Qemb, w.Q128, w);

    // ------- context-query attention -------
    float* Scq = w.uni;
    float* S1  = w.uni + 700000;
    float* S2  = w.uni + 1400000;
    float* s0  = w.uni + 2100000;
    float* s1v = w.uni + 2120000;
    float* Cw  = w.uni + 2200000;
    float* Acq = w.uni + 4000000;
    float* Bt  = w.uni + 6000000;
    float* Bv  = w.uni + 11200000;

    k_scale_cols<<<nblk(nC128), 256, 0, stream>>>(w.C128, w4mul, Cw, nC128);
    gemm_full(stream, Cw, w.Q128, Scq, nullptr, PARA, QUES, DM,
              DM, DM, QUES,
              (long)PARA * DM, 0, (long)QUES * DM, 0, (long)PARA * QUES, 0,
              1, B_N, 1, 0, 1.0f);
    k_dotw<<<nblk(B_N * PARA), 256, 0, stream>>>(w.C128, w4c, s0, (long)B_N * PARA, DM);
    k_dotw<<<nblk(B_N * QUES), 256, 0, stream>>>(w.Q128, w4q, s1v, (long)B_N * QUES, DM);
    long nS = (long)B_N * PARA * QUES;
    k_addS<<<nblk(nS), 256, 0, stream>>>(Scq, s0, s1v, cq_bias, PARA, QUES, nS);
    k_softmax_rows<<<B_N * PARA, 256, 0, stream>>>(Scq, S1, w.qmask, PARA, QUES);
    k_softmax_axis1<<<B_N * QUES, 256, 0, stream>>>(Scq, S2, w.cmask, PARA, QUES);
    gemm_full(stream, S1, w.Q128, Acq, nullptr, PARA, DM, QUES,
              QUES, DM, DM,
              (long)PARA * QUES, 0, (long)QUES * DM, 0, (long)PARA * DM, 0,
              1, B_N, 0, 0, 1.0f);
    gemm_full(stream, S1, S2, Bt, nullptr, PARA, PARA, QUES,
              QUES, QUES, PARA,
              (long)PARA * QUES, 0, (long)PARA * QUES, 0, (long)PARA * PARA, 0,
              1, B_N, 1, 0, 1.0f);
    gemm_full(stream, Bt, w.C128, Bv, nullptr, PARA, DM, PARA,
              PARA, DM, DM,
              (long)PARA * PARA, 0, (long)PARA * DM, 0, (long)PARA * DM, 0,
              1, B_N, 0, 0, 1.0f);
    long n512 = (long)B_N * PARA * 4 * DM;
    k_concat_cq<<<nblk(n512), 256, 0, stream>>>(w.C128, Acq, Bv, w.cq512, n512);

    // ------- resizer (dsconv, no relu) -------
    float* t512 = w.uni;
    k_dwconv<<<nblk(n512), 256, 0, stream>>>(w.cq512, rdw, rdb, t512, PARA, 4 * DM, 5, 0, n512);
    gemm2(stream, t512, rpw, w.M1, rpb, B_N * PARA, DM, 4 * DM, 1, 0);

    // ------- decoder stacks (shared weights, 3 passes) -------
    for (int i = 0; i < 7; ++i) run_enc(stream, dec[i], w.cmask, w.M1, w.M1, w);
    hipMemcpyAsync(w.M2, w.M1, sizeof(float) * nC128, hipMemcpyDeviceToDevice, stream);
    for (int i = 0; i < 7; ++i) run_enc(stream, dec[i], w.cmask, w.M2, w.M2, w);
    hipMemcpyAsync(w.M3, w.M2, sizeof(float) * nC128, hipMemcpyDeviceToDevice, stream);
    for (int i = 0; i < 7; ++i) run_enc(stream, dec[i], w.cmask, w.M3, w.M3, w);

    // ------- pointer heads + masked log-softmax -------
    float* pt1 = w.uni;
    float* pt2 = w.uni + 20000;
    k_ptr<<<nblk(B_N * PARA), 256, 0, stream>>>(w.M1, w.M2, w1, b1, pt1, (long)B_N * PARA, DM);
    k_ptr<<<nblk(B_N * PARA), 256, 0, stream>>>(w.M1, w.M3, w2, b2, pt2, (long)B_N * PARA, DM);
    float* outp = (float*)d_out;
    k_logsoftmax<<<B_N, 256, 0, stream>>>(pt1, w.cmask, outp, PARA);
    k_logsoftmax<<<B_N, 256, 0, stream>>>(pt2, w.cmask, outp + (long)B_N * PARA, PARA);
}
